// WindowedMultiHeadSelfAttention_48868137893995
// MI455X (gfx1250) — compile-verified, matrix operand order checked by reading, not run on hardware
//
#include <hip/hip_runtime.h>

typedef _Float16 half_t;
typedef __attribute__((ext_vector_type(16))) _Float16 v16h;
typedef __attribute__((ext_vector_type(8)))  _Float16 v8h;
typedef __attribute__((ext_vector_type(8)))  float    v8f;

#define WIN_N   49
#define NPAD    64
#define CDIM    256
#define NHEADS  8
#define HD      32
#define NWIN    64

// ---- LDS layout (bytes) ----
#define XS_OFF   0         // xs [64][256] f16 (32 KB); aliased by S f32 [8][16][64] in phase 3
#define QS_OFF   32768     // qs [8][64][32] f16 (32 KB); staging aliases qs+part of ks in phase 1
#define KS_OFF   65536     // ks [8][64][32] f16 (32 KB)
#define VS_OFF   98304     // vs [8][32][64] f16 transposed (32 KB)
#define AO_OFF   131072    // ao [64][256] f16 (32 KB)
#define SP_OFF   163840    // probs f16 [8][16][64] (16 KB)
#define SMEM_TOTAL 180224

// ---- workspace layout (bytes) ----
#define WS_QKVW   0                     // packed qkv_w f16: 48*8*32*16 halves = 393216 B
#define WS_PROJW  393216                // packed proj_w f16: 16*8*32*16 halves = 131072 B
#define WS_BIAS   524288                // gathered bias f32 [8][49][49] = 76832 B

// Load a 16x32 f16 A/B fragment from a row-major half array in LDS.
// ISA 7.12.2: lanes 0-15 row=lane hold K {0..7,16..23}, lanes 16-31 same rows
// hold K {8..15,24..31}. B fragments use the same pattern with "row" = column
// index of B read from the transposed source.
__device__ __forceinline__ v16h load_frag_rm(const half_t* base, int row, int ld,
                                             int kt, int lane) {
  int kb = (lane & 16) ? 8 : 0;
  const half_t* p = base + row * ld + kt * 32;
  v8h lo = *(const v8h*)(p + kb);
  v8h hi = *(const v8h*)(p + 16 + kb);
  return __builtin_shufflevector(lo, hi, 0,1,2,3,4,5,6,7,8,9,10,11,12,13,14,15);
}

// Pre-pack weight matrix w[K=256][ncols] into per-lane B-fragment blobs:
// out[((nt*8 + kt)*32 + lane)*16 + j]
__global__ void pack_w_kernel(const float* __restrict__ w, half_t* __restrict__ o,
                              int ntiles, int ncols) {
  int idx = blockIdx.x * 256 + threadIdx.x;
  int total = ntiles * 8 * 32 * 16;
  if (idx >= total) return;
  int j    = idx & 15;
  int lane = (idx >> 4) & 31;
  int kt   = (idx >> 9) & 7;
  int nt   = idx >> 12;
  int n  = nt * 16 + (lane & 15);
  int kb = (lane & 16) ? 8 : 0;
  int K  = kt * 32 + ((j < 8) ? (kb + j) : (16 + kb + (j - 8)));
  o[idx] = (half_t)w[K * ncols + n];
}

// biasM[h][i][j] = bias_table[rel_idx[i*49+j]*8 + h]
__global__ void bias_gather_kernel(const float* __restrict__ bt,
                                   const int* __restrict__ ridx,
                                   float* __restrict__ o) {
  int idx = blockIdx.x * 256 + threadIdx.x;
  if (idx >= NHEADS * WIN_N * WIN_N) return;
  int h   = idx / (WIN_N * WIN_N);
  int rem = idx % (WIN_N * WIN_N);
  o[idx] = bt[ridx[rem] * NHEADS + h];
}

__global__ __launch_bounds__(256, 1)
void swin_wmsa_kernel(const float* __restrict__ x,
                      const float* __restrict__ mask,
                      const float* __restrict__ qkv_b,
                      const float* __restrict__ proj_b,
                      const half_t* __restrict__ wqkv_p,
                      const half_t* __restrict__ wproj_p,
                      const float* __restrict__ biasM,
                      float* __restrict__ out) {
  extern __shared__ char sm[];
  half_t* xs = (half_t*)(sm + XS_OFF);
  half_t* qs = (half_t*)(sm + QS_OFF);
  half_t* ks = (half_t*)(sm + KS_OFF);
  half_t* vs = (half_t*)(sm + VS_OFF);
  half_t* ao = (half_t*)(sm + AO_OFF);
  half_t* sp = (half_t*)(sm + SP_OFF);
  float*  sb = (float*)(sm + XS_OFF);   // aliases xs (dead after phase 2)

  const int tid  = threadIdx.x;
  const int lane = tid & 31;
  const int wv   = tid >> 5;
  const int b    = blockIdx.x;
  const float* xb = x + (size_t)b * WIN_N * CDIM;

  // hint: packed weights into cache early
  __builtin_prefetch(wqkv_p + (size_t)wv * 6 * 8 * 512, 0, 1);

  // ------- Phase 1: async DMA x[b] (f32) -> LDS staging, then cvt to f16 ----
  // staging region aliases qs (written only after the conversion barrier)
  for (int idx = tid; idx < (WIN_N * CDIM) / 4; idx += 256) {
    unsigned int ldsoff = (unsigned int)(QS_OFF + idx * 16);
    unsigned long long ga = (unsigned long long)(uintptr_t)(xb + idx * 4);
    asm volatile("global_load_async_to_lds_b128 %0, %1, off"
                 :: "v"(ldsoff), "v"(ga) : "memory");
  }
  asm volatile("s_wait_asynccnt 0x0" ::: "memory");
  __syncthreads();
  {
    const float* stf = (const float*)(sm + QS_OFF);
    for (int r = 0; r < NPAD; ++r) {
      float v = (r < WIN_N) ? stf[r * CDIM + tid] : 0.0f;
      xs[r * CDIM + tid] = (half_t)v;
    }
  }
  __syncthreads();

  // ---------------- Phase 2: QKV = x @ Wqkv + b ----------------------------
  const float scale = 0.17677669529663687f;  // 32^-0.5
  for (int mt = 0; mt < 4; ++mt) {
    v16h afr[8];
#pragma unroll
    for (int kt = 0; kt < 8; ++kt)
      afr[kt] = load_frag_rm(xs, mt * 16 + (lane & 15), CDIM, kt, lane);
    for (int t6 = 0; t6 < 6; ++t6) {
      int nt = wv * 6 + t6;
      v8f acc = {};
#pragma unroll
      for (int kt = 0; kt < 8; ++kt) {
        v16h bf = *(const v16h*)(wqkv_p + ((size_t)(nt * 8 + kt) * 32 + lane) * 16);
        acc = __builtin_amdgcn_wmma_f32_16x16x32_f16(false, afr[kt], false, bf,
                                                     (short)0, acc, false, false);
      }
      int l     = lane & 15;
      int sec   = nt >> 4;            // uniform: 0=q 1=k 2=v
      int h     = (nt & 15) >> 1;     // uniform head
      int d     = ((nt & 1) << 4) + l;
      float bias = qkv_b[nt * 16 + l];
      int rbase = (lane & 16) ? 8 : 0;
#pragma unroll
      for (int r = 0; r < 8; ++r) {
        int m = mt * 16 + rbase + r;
        float vv = acc[r] + bias;
        if (sec == 0)      qs[(h * NPAD + m) * HD + d] = (half_t)(vv * scale);
        else if (sec == 1) ks[(h * NPAD + m) * HD + d] = (half_t)vv;
        else               vs[(h * HD + d) * NPAD + m] = (half_t)vv;  // transposed
      }
    }
  }
  __syncthreads();

  // ---------------- Phase 3: attention, one head per wave ------------------
  {
    const int h = wv;
    const half_t* qh = qs + h * NPAD * HD;
    const half_t* kh = ks + h * NPAD * HD;
    const half_t* vh = vs + h * HD * NPAD;
    float*  sbw = sb + wv * 16 * NPAD;
    half_t* spw = sp + wv * 16 * NPAD;
    const float* mwin = mask + (size_t)(b & (NWIN - 1)) * WIN_N * WIN_N;
    const float* bh   = biasM + (size_t)h * WIN_N * WIN_N;

    // mt-invariant fragments: K (4 tiles), V (2 dt x 2 kt)
    v16h kfr[4], vfr[4];
#pragma unroll
    for (int ntj = 0; ntj < 4; ++ntj)
      kfr[ntj] = load_frag_rm(kh, ntj * 16 + (lane & 15), HD, 0, lane);
#pragma unroll
    for (int q2 = 0; q2 < 4; ++q2)   // q2 = dt*2 + kt
      vfr[q2] = load_frag_rm(vh, (q2 >> 1) * 16 + (lane & 15), NPAD, q2 & 1, lane);

    for (int mt = 0; mt < 4; ++mt) {
      // S = q @ k^T  (K=32 -> single WMMA per 16x16 tile)
      v16h aq = load_frag_rm(qh, mt * 16 + (lane & 15), HD, 0, lane);
#pragma unroll
      for (int ntj = 0; ntj < 4; ++ntj) {
        v8f s = {};
        s = __builtin_amdgcn_wmma_f32_16x16x32_f16(false, aq, false, kfr[ntj],
                                                   (short)0, s, false, false);
        int j     = ntj * 16 + (lane & 15);
        int rbase = (lane & 16) ? 8 : 0;
#pragma unroll
        for (int r = 0; r < 8; ++r) {
          int il = rbase + r;
          int i  = mt * 16 + il;
          bool ok = (i < WIN_N) & (j < WIN_N);
          int ic = i < WIN_N ? i : WIN_N - 1;
          int jc = j < WIN_N ? j : WIN_N - 1;
          float val = s[r] + bh[ic * WIN_N + jc] + mwin[ic * WIN_N + jc];
          sbw[il * NPAD + j] = ok ? val : -1e30f;
        }
      }
      __syncthreads();

      // row softmax: 2 lanes per row, 32 cols each, combine via shfl_xor
      {
        int srow = lane >> 1;
        int jb   = (lane & 1) * 32;
        float* row = sbw + srow * NPAD;
        float mx = -1e30f;
#pragma unroll
        for (int j = 0; j < 32; ++j) mx = fmaxf(mx, row[jb + j]);
        mx = fmaxf(mx, __shfl_xor(mx, 1, 32));
        float sum = 0.0f;
#pragma unroll
        for (int j = 0; j < 32; ++j) sum += __expf(row[jb + j] - mx);
        sum += __shfl_xor(sum, 1, 32);
        float inv = 1.0f / sum;
#pragma unroll
        for (int j = 0; j < 32; ++j)
          spw[srow * NPAD + jb + j] = (half_t)(__expf(row[jb + j] - mx) * inv);
      }
      __syncthreads();

      // O = P @ V  (K=64 -> 2 WMMA k-steps; N=32 -> 2 tiles)
      v16h ap0 = load_frag_rm(spw, (lane & 15), NPAD, 0, lane);
      v16h ap1 = load_frag_rm(spw, (lane & 15), NPAD, 1, lane);
#pragma unroll
      for (int dt = 0; dt < 2; ++dt) {
        v8f acc = {};
        acc = __builtin_amdgcn_wmma_f32_16x16x32_f16(false, ap0, false, vfr[dt * 2],
                                                     (short)0, acc, false, false);
        acc = __builtin_amdgcn_wmma_f32_16x16x32_f16(false, ap1, false, vfr[dt * 2 + 1],
                                                     (short)0, acc, false, false);
        int col   = h * HD + dt * 16 + (lane & 15);
        int rbase = (lane & 16) ? 8 : 0;
#pragma unroll
        for (int r = 0; r < 8; ++r) {
          int m = mt * 16 + rbase + r;
          ao[m * CDIM + col] = (half_t)acc[r];
        }
      }
      __syncthreads();
    }
  }

  // ---------------- Phase 4: out = ao @ Wproj + b --------------------------
  for (int mt = 0; mt < 4; ++mt) {
    v16h afr[8];
#pragma unroll
    for (int kt = 0; kt < 8; ++kt)
      afr[kt] = load_frag_rm(ao, mt * 16 + (lane & 15), CDIM, kt, lane);
    for (int t2 = 0; t2 < 2; ++t2) {
      int nt = wv * 2 + t2;
      v8f acc = {};
#pragma unroll
      for (int kt = 0; kt < 8; ++kt) {
        v16h bf = *(const v16h*)(wproj_p + ((size_t)(nt * 8 + kt) * 32 + lane) * 16);
        acc = __builtin_amdgcn_wmma_f32_16x16x32_f16(false, afr[kt], false, bf,
                                                     (short)0, acc, false, false);
      }
      int col   = nt * 16 + (lane & 15);
      float pb  = proj_b[col];
      int rbase = (lane & 16) ? 8 : 0;
#pragma unroll
      for (int r = 0; r < 8; ++r) {
        int m = mt * 16 + rbase + r;
        if (m < WIN_N)
          __builtin_nontemporal_store(acc[r] + pb,
              &out[((size_t)b * WIN_N + m) * CDIM + col]);
      }
    }
  }
}

extern "C" void kernel_launch(void* const* d_in, const int* in_sizes, int n_in,
                              void* d_out, int out_size, void* d_ws, size_t ws_size,
                              hipStream_t stream) {
  const float* x        = (const float*)d_in[0];
  const float* mask     = (const float*)d_in[1];
  const float* qkv_w    = (const float*)d_in[2];
  const float* qkv_b    = (const float*)d_in[3];
  const float* bias_tab = (const float*)d_in[4];
  const float* proj_w   = (const float*)d_in[5];
  const float* proj_b   = (const float*)d_in[6];
  const int*   rel_idx  = (const int*)d_in[7];
  float* out = (float*)d_out;

  half_t* wqkv_p  = (half_t*)((char*)d_ws + WS_QKVW);
  half_t* wproj_p = (half_t*)((char*)d_ws + WS_PROJW);
  float*  biasM   = (float*)((char*)d_ws + WS_BIAS);

  // pack weights into WMMA B-fragment layout (f16)
  pack_w_kernel<<<(48 * 8 * 32 * 16 + 255) / 256, 256, 0, stream>>>(qkv_w, wqkv_p, 48, 768);
  pack_w_kernel<<<(16 * 8 * 32 * 16 + 255) / 256, 256, 0, stream>>>(proj_w, wproj_p, 16, 256);
  bias_gather_kernel<<<(NHEADS * WIN_N * WIN_N + 255) / 256, 256, 0, stream>>>(bias_tab, rel_idx, biasM);

  swin_wmsa_kernel<<<4096, 256, SMEM_TOTAL, stream>>>(x, mask, qkv_b, proj_b,
                                                      wqkv_p, wproj_p, biasM, out);
}
